// CNNFusing_68436008895088
// MI455X (gfx1250) — matrix operand order repaired and checked by reading, not compiled, not run on hardware
//
#include <hip/hip_runtime.h>

constexpr int Hdim = 128;

typedef __attribute__((ext_vector_type(16))) __bf16 v16bf;
typedef __attribute__((ext_vector_type(8)))  float  v8f;

// DPP row-rotate-right add: sums across a 16-lane row in 4 steps, no LDS.
#define ROW_ROR_ADD(v, n)                                                  \
  ((v) + __builtin_bit_cast(float, __builtin_amdgcn_update_dpp(            \
             0, __builtin_bit_cast(int, (v)), 0x120 + (n), 0xF, 0xF, true)))

// ---------------------------------------------------------------------------
// Kernel 0: exclusive prefix sum of seq_len -> token offsets (ragged-safe)
// ---------------------------------------------------------------------------
__global__ void scan_offsets_kernel(const int* __restrict__ seq_len,
                                    int* __restrict__ off, int nB) {
  __shared__ int chunk[256];
  const int tid = threadIdx.x;
  const int per = (nB + 255) / 256;
  const int base = tid * per;
  int s = 0;
  for (int i = 0; i < per; ++i) {
    int b = base + i;
    if (b < nB) s += seq_len[b];
  }
  chunk[tid] = s;
  __syncthreads();
  if (tid == 0) {
    int run = 0;
    for (int i = 0; i < 256; ++i) { int v = chunk[i]; chunk[i] = run; run += v; }
  }
  __syncthreads();
  int run = chunk[tid];
  for (int i = 0; i < per; ++i) {
    int b = base + i;
    if (b < nB) { off[b] = run; run += seq_len[b]; }
  }
}

// ---------------------------------------------------------------------------
// Kernel 1: one workgroup (8 wave32s) per session.
//   hidden = max(intra, inter)            (fp32, prefetched + double-buffered)
//   z      = sigmoid(hidden @ W2^T + u)   (bf16 WMMA, f32 accumulate)
//   alpha  = z . q_w + q_b                (DPP row reduction, v_rcp sigmoid)
//   s_g    = sum_t alpha_t * hidden_t     (fp32 pooling in registers)
//   out    = W3 @ [v_n ; s_g] + b3        (fp32)
// ---------------------------------------------------------------------------
__global__ __launch_bounds__(256)
void session_attn_kernel(const float* __restrict__ intra,
                         const float* __restrict__ inter,
                         const float* __restrict__ W1,
                         const float* __restrict__ b1,
                         const float* __restrict__ W2,
                         const float* __restrict__ b2,
                         const float* __restrict__ q_w,
                         const float* __restrict__ q_b,
                         const float* __restrict__ W3,
                         const float* __restrict__ b3,
                         const int*   __restrict__ seq_len,
                         const int*   __restrict__ off,
                         float* __restrict__ out) {
  constexpr int SF = Hdim + 4;   // fp32 tile stride (bank-conflict padding)
  constexpr int SB = Hdim + 8;   // bf16 tile stride (32-bit pair padding)
  __shared__ float  vn[Hdim];
  __shared__ float  uvec[Hdim];
  __shared__ float  hidf[2][16][SF];   // double-buffered fused tile, fp32
  __shared__ __bf16 hidb[2][16][SB];   // double-buffered fused tile, bf16
  __shared__ float  alpha_part[8][16];
  __shared__ float  alpha[16];
  __shared__ float  sgbuf[Hdim];

  const int b    = blockIdx.x;
  const int tid  = threadIdx.x;
  const int lane = tid & 31;
  const int wave = tid >> 5;
  const int Lb   = seq_len[b];
  const long t0  = (long)off[b];

  // --- v_n: max-fused last token of this session --------------------------
  if (tid < Hdim) {
    long tl = (t0 + (Lb > 0 ? Lb : 1) - 1) * Hdim + tid;
    vn[tid] = fmaxf(intra[tl], inter[tl]);
  }
  __syncthreads();

  // --- u[n] = b1[n] + b2[n] + sum_k W1[n][k] * v_n[k] ---------------------
  if (tid < Hdim) {
    float acc = b1[tid] + b2[tid];
    const float* wrow = W1 + (long)tid * Hdim;
#pragma unroll 4
    for (int k = 0; k < Hdim; ++k) acc = fmaf(wrow[k], vn[k], acc);
    uvec[tid] = acc;
  }

  // --- this wave's 16-column slice of W2^T as bf16 B fragments ------------
  // B[k][n] = W2[n][k]; element e of frag kk: K = 32*kk + 16*(lane/16) + e
  const int ncol  = wave * 16 + (lane & 15);
  const int khalf = (lane >> 4) * 16;
  v16bf bfrag[4];
  {
    const float* w2row = W2 + (long)ncol * Hdim;
#pragma unroll
    for (int kk = 0; kk < 4; ++kk)
#pragma unroll
      for (int e = 0; e < 16; ++e)
        bfrag[kk][e] = (__bf16)w2row[kk * 32 + khalf + e];
  }

  __syncthreads();
  const float u_n  = uvec[ncol];
  const float qw_n = q_w[ncol];
  const float qb   = q_b[0];

  float sg_acc = 0.0f;                 // thread tid<128 owns column tid of s_g
  const int ntiles = (Lb + 15) >> 4;

  // staging geometry: each thread owns 8 consecutive columns of one row
  const int sm = tid >> 4;             // 0..15 row within tile
  const int sc = (tid & 15) * 8;       // column start

  // --- prefetch tile 0 into registers (fused max) -------------------------
  float pv[8];
  {
    const bool valid = (0 < ntiles) && (sm < Lb);
    if (valid) {
      const long gbase = (t0 + sm) * (long)Hdim + sc;
      const float4 ia = *(const float4*)(intra + gbase);
      const float4 ib = *(const float4*)(intra + gbase + 4);
      const float4 ra = *(const float4*)(inter + gbase);
      const float4 rb = *(const float4*)(inter + gbase + 4);
      pv[0] = fmaxf(ia.x, ra.x); pv[1] = fmaxf(ia.y, ra.y);
      pv[2] = fmaxf(ia.z, ra.z); pv[3] = fmaxf(ia.w, ra.w);
      pv[4] = fmaxf(ib.x, rb.x); pv[5] = fmaxf(ib.y, rb.y);
      pv[6] = fmaxf(ib.z, rb.z); pv[7] = fmaxf(ib.w, rb.w);
    } else {
#pragma unroll
      for (int j = 0; j < 8; ++j) pv[j] = 0.0f;
    }
  }

  for (int tile = 0; tile < ntiles; ++tile) {
    const int cur = tile & 1;

    // --- commit prefetched tile to LDS (fp32 + bf16 copies) ---------------
#pragma unroll
    for (int j = 0; j < 8; ++j) {
      hidf[cur][sm][sc + j] = pv[j];
      hidb[cur][sm][sc + j] = (__bf16)pv[j];
    }
    __syncthreads();

    // --- prefetch next tile (global loads overlap WMMA + epilogue) --------
    if (tile + 1 < ntiles) {
      const int row = (tile + 1) * 16 + sm;
      if (row < Lb) {
        const long gbase = (t0 + row) * (long)Hdim + sc;
        const float4 ia = *(const float4*)(intra + gbase);
        const float4 ib = *(const float4*)(intra + gbase + 4);
        const float4 ra = *(const float4*)(inter + gbase);
        const float4 rb = *(const float4*)(inter + gbase + 4);
        pv[0] = fmaxf(ia.x, ra.x); pv[1] = fmaxf(ia.y, ra.y);
        pv[2] = fmaxf(ia.z, ra.z); pv[3] = fmaxf(ia.w, ra.w);
        pv[4] = fmaxf(ib.x, rb.x); pv[5] = fmaxf(ib.y, rb.y);
        pv[6] = fmaxf(ib.z, rb.z); pv[7] = fmaxf(ib.w, rb.w);
      } else {
#pragma unroll
        for (int j = 0; j < 8; ++j) pv[j] = 0.0f;
      }
    }

    // --- load ALL A fragments first (8x ds_load_b128, one wait), then -----
    // --- run the 4 WMMAs back-to-back (K=128) -----------------------------
    const int mrow = lane & 15;
    v16bf afrag[4];
#pragma unroll
    for (int kk = 0; kk < 4; ++kk)
#pragma unroll
      for (int e = 0; e < 16; ++e) {
        int k = kk * 32 + (lane >> 4) * 16 + e;
        afrag[kk][e] = hidb[cur][mrow][k];
      }
    v8f acc = {};
#pragma unroll
    for (int kk = 0; kk < 4; ++kk)
      acc = __builtin_amdgcn_wmma_f32_16x16x32_bf16(
          false, afrag[kk], false, bfrag[kk], (short)0, acc, false, false);

    // --- batched sigmoid epilogue: all 8 exp/rcp chains pipelined ---------
    float p[8];
#pragma unroll
    for (int r = 0; r < 8; ++r) {
      float x = acc[r] + u_n;                       // row m = r + 8*(lane/16)
      p[r] = qw_n * __builtin_amdgcn_rcpf(1.0f + __expf(-x));
    }
    // DPP row reductions over the 16 n-lanes, all rows interleaved
#pragma unroll
    for (int r = 0; r < 8; ++r) p[r] = ROW_ROR_ADD(p[r], 1);
#pragma unroll
    for (int r = 0; r < 8; ++r) p[r] = ROW_ROR_ADD(p[r], 2);
#pragma unroll
    for (int r = 0; r < 8; ++r) p[r] = ROW_ROR_ADD(p[r], 4);
#pragma unroll
    for (int r = 0; r < 8; ++r) p[r] = ROW_ROR_ADD(p[r], 8);
    if ((lane & 15) == 0) {                          // one contiguous 32B run
#pragma unroll
      for (int r = 0; r < 8; ++r)
        alpha_part[wave][(lane >> 4) * 8 + r] = p[r];
    }
    __syncthreads();
    if (tid < 16) {                    // combine the 8 waves' column tiles
      float a = qb;
#pragma unroll
      for (int w = 0; w < 8; ++w) a += alpha_part[w][tid];
      alpha[tid] = a;
    }
    __syncthreads();

    // --- s_g pooling: exact fp32 hidden * alpha ---------------------------
    if (tid < Hdim) {
#pragma unroll
      for (int m = 0; m < 16; ++m)
        sg_acc = fmaf(alpha[m], hidf[cur][m][tid], sg_acc);
    }
    // next iteration writes the OTHER tile buffer; alpha_part rewrites are
    // ordered by the barrier right after the LDS commit above.
  }

  __syncthreads();
  if (tid < Hdim) sgbuf[tid] = sg_acc;
  __syncthreads();

  // --- h_s = W3 @ [v_n ; s_g] + b3 ----------------------------------------
  if (tid < Hdim) {
    const float* w3row = W3 + (long)tid * (2 * Hdim);
    float o = b3[tid];
#pragma unroll 4
    for (int k = 0; k < Hdim; ++k) o = fmaf(w3row[k], vn[k], o);
#pragma unroll 4
    for (int k = 0; k < Hdim; ++k) o = fmaf(w3row[Hdim + k], sgbuf[k], o);
    out[(long)b * Hdim + tid] = o;
  }
}

// ---------------------------------------------------------------------------
extern "C" void kernel_launch(void* const* d_in, const int* in_sizes, int n_in,
                              void* d_out, int out_size, void* d_ws, size_t ws_size,
                              hipStream_t stream) {
  const float* intra = (const float*)d_in[0];
  const float* inter = (const float*)d_in[1];
  const float* W1    = (const float*)d_in[2];
  const float* b1    = (const float*)d_in[3];
  const float* W2    = (const float*)d_in[4];
  const float* b2    = (const float*)d_in[5];
  const float* q_w   = (const float*)d_in[6];
  const float* q_b   = (const float*)d_in[7];
  const float* W3    = (const float*)d_in[8];
  const float* b3    = (const float*)d_in[9];
  const int* seq_len = (const int*)d_in[10];
  const int nB = in_sizes[10];

  int* off = (int*)d_ws;   // exclusive token offsets per session

  scan_offsets_kernel<<<1, 256, 0, stream>>>(seq_len, off, nB);
  session_attn_kernel<<<nB, 256, 0, stream>>>(intra, inter, W1, b1, W2, b2,
                                              q_w, q_b, W3, b3, seq_len, off,
                                              (float*)d_out);
}
